// ECGNNEdgePredictor_80315888435896
// MI455X (gfx1250) — compile-verified, matrix operand order checked by reading, not run on hardware
//
#include <hip/hip_runtime.h>
#include <hip/hip_bf16.h>

// ---------------------------------------------------------------------------
// ECGNN edge predictor, fused bf16-WMMA pipeline for gfx1250 (MI455X).
//   h = x@Wp+bp
//   5x: hidden = relu([h_src|h_dst]@W1+b1)           (WMMA, [E,128]x[128,32])
//       msg    = P @ W2resh                          (WMMA, [E,2112]x[2112,64])
//                P[e, k*64+h] = hidden[e,k]*h_src[e,h], bias rows folded in.
//                W2resh staged in LDS per workgroup via the Tensor Data Mover.
//       agg    = scatter-mean(msg, dst)              (f32 global atomics)
//       h      = relu(BN(agg + h@Wr + br)) + h       (WMMA + fused epilogue)
//   out = relu([h_u|h_v]@Wm1+bm1)@Wm2+bm2            (WMMA + shuffle reduce)
// Weights pre-swizzled once per launch into per-lane WMMA-fragment order so
// every B fragment is two contiguous 128-bit loads (global or ds).
// ---------------------------------------------------------------------------

#define HID 64
#define MSG_LDS_BYTES (66 * 4 * 32 * 16 * 2)   // 270336 B: full W2resh layer

typedef unsigned short u16t;
typedef __attribute__((ext_vector_type(8)))  unsigned short u16x8;
typedef __attribute__((ext_vector_type(16))) __bf16        v16bf;
typedef __attribute__((ext_vector_type(8)))  float         v8f;
typedef __attribute__((ext_vector_type(4)))  unsigned int  u32x4;
typedef __attribute__((ext_vector_type(8)))  int           i32x8;
typedef __attribute__((ext_vector_type(4)))  int           i32x4;

union F16Frag {
    u16x8 h[2];
    u16t  us[16];
    v16bf bf;
};

__device__ __forceinline__ float bf2f(u16t b) {
    union { unsigned u; float f; } x; x.u = ((unsigned)b) << 16; return x.f;
}
__device__ __forceinline__ u16t f2bf(float f) {
    union { float f; unsigned u; } x; x.f = f;
    unsigned u = x.u;
    return (u16t)((u + 0x7FFFu + ((u >> 16) & 1u)) >> 16);
}

__device__ __forceinline__ v8f wmma_bf16(v16bf a, v16bf b, v8f c) {
#if defined(__gfx1250__)
    return __builtin_amdgcn_wmma_f32_16x16x32_bf16(
        /*neg_a=*/false, a, /*neg_b=*/false, b,
        /*c_mod=*/(short)0, c, /*reuse_a=*/false, /*reuse_b=*/false);
#else
    (void)a; (void)b; return c;
#endif
}

// 16-bit A-matrix fragment (16x32 MxK): lane<16 holds K={kb..kb+7, kb+16..kb+23}
// of row (lane&15), kb=0; lane>=16 same row, kb=8.  rp points at a bf16 row.
__device__ __forceinline__ v16bf load_frag_row(const u16t* rp, int kbase) {
    F16Frag f;
    f.h[0] = *(const u16x8*)(rp + kbase);
    f.h[1] = *(const u16x8*)(rp + kbase + 16);
    return f.bf;
}

// Swizzled-weight B fragment: 16 contiguous bf16 per (kc,nt,lane).
__device__ __forceinline__ v16bf load_frag_w(const u16t* Ws, int kc, int NT, int nt, int lane) {
    const u16t* p = Ws + (size_t)(((kc * NT + nt) * 32 + lane) * 16);
    F16Frag f;
    f.h[0] = *(const u16x8*)(p);
    f.h[1] = *(const u16x8*)(p + 8);
    return f.bf;
}

// a = hs * splat(hbits)   (packed bf16 multiply -> v_pk_mul_bf16)
__device__ __forceinline__ v16bf scale_frag(v16bf hs, u16t hbits) {
    union { u16t u; __bf16 b; } c; c.u = hbits;
    return hs * c.b;
}

// ---------------------------------------------------------------------------
// Weight pre-swizzle: layout [kc][nt][lane][j], kk = kc*32 + kb + run(j),
// n = nt*16 + (lane&15).  W2 gets the P-reshape: row kk<2048 -> W2[k][h*64+n]
// with k=kk>>6, h=kk&63; rows 2048..2111 -> b2[h*64+n] (bias folded into GEMM).
// ---------------------------------------------------------------------------
__device__ __forceinline__ void decode_frag(int r, int NT, int& kk, int& n) {
    int j = r & 15, lane = (r >> 4) & 31, t = r >> 9;
    int nt = t % NT, kc = t / NT;
    int kbase = (lane < 16) ? 0 : 8;
    kk = kc * 32 + kbase + ((j < 8) ? j : 16 + (j - 8));
    n  = nt * 16 + (lane & 15);
}

__global__ void k_prep(const float* __restrict__ W1, const float* __restrict__ W2,
                       const float* __restrict__ b2, const float* __restrict__ Wr,
                       const float* __restrict__ Wm1,
                       u16t* __restrict__ W1s, u16t* __restrict__ W2s,
                       u16t* __restrict__ Wrs, u16t* __restrict__ Wm1s) {
    const int N1 = 5 * 4096;      // W1: 5 layers, [4 kc][2 nt][32][16]
    const int N2 = 5 * 135168;    // W2: 5 layers, [66 kc][4 nt][32][16]
    const int NR = 5 * 4096;      // Wr: 5 layers, [2 kc][4 nt][32][16]
    const int NM = 8192;          // Wm1: [4 kc][4 nt][32][16]
    int tot = N1 + N2 + NR + NM;
    for (int idx = blockIdx.x * blockDim.x + threadIdx.x; idx < tot;
         idx += gridDim.x * blockDim.x) {
        int kk, n;
        if (idx < N1) {
            int l = idx / 4096, r = idx % 4096;
            decode_frag(r, 2, kk, n);
            W1s[idx] = f2bf(W1[l * (128 * 32) + kk * 32 + n]);
        } else if (idx < N1 + N2) {
            int s = idx - N1;
            int l = s / 135168, r = s % 135168;
            decode_frag(r, 4, kk, n);
            float v;
            if (kk < 2048) {
                int k = kk >> 6, hh = kk & 63;
                v = W2[(size_t)l * 131072 + k * 4096 + hh * 64 + n];
            } else {
                v = b2[l * 4096 + (kk - 2048) * 64 + n];
            }
            W2s[s] = f2bf(v);
        } else if (idx < N1 + N2 + NR) {
            int s = idx - N1 - N2;
            int l = s / 4096, r = s % 4096;
            decode_frag(r, 4, kk, n);
            Wrs[s] = f2bf(Wr[l * 4096 + kk * 64 + n]);
        } else {
            int s = idx - N1 - N2 - NR;
            decode_frag(s, 4, kk, n);
            Wm1s[s] = f2bf(Wm1[kk * 64 + n]);
        }
    }
}

// ---------------------------------------------------------------------------
__global__ void k_proj(const float* __restrict__ x, const float* __restrict__ Wp,
                       const float* __restrict__ bp, float* __restrict__ h,
                       u16t* __restrict__ hbf, int N) {
    int i = blockIdx.x * blockDim.x + threadIdx.x;
    if (i >= N * HID) return;
    int node = i >> 6, c = i & 63;
    const float* xr = x + node * 8;
    float v = bp[c];
#pragma unroll
    for (int k = 0; k < 8; k++) v += xr[k] * Wp[k * HID + c];
    h[i] = v;
    hbf[i] = f2bf(v);
}

__global__ void k_zero(float* __restrict__ p, int n) {
    int i = blockIdx.x * blockDim.x + threadIdx.x;
    if (i < n) p[i] = 0.f;
}

__global__ void k_deg(const int* __restrict__ dst, float* __restrict__ deg, int E) {
    int e = blockIdx.x * blockDim.x + threadIdx.x;
    if (e < E) atomicAdd(&deg[dst[e]], 1.0f);
}

// ---------------------------------------------------------------------------
// hidden = relu([h_src|h_dst] @ W1 + b1) : [E,128]x[128,32], wave per 16 edges
// ---------------------------------------------------------------------------
__global__ void k_edge_hidden(const u16t* __restrict__ hbf,
                              const int* __restrict__ src, const int* __restrict__ dst,
                              const u16t* __restrict__ W1s, const float* __restrict__ b1,
                              u16t* __restrict__ hid, int E, int ntiles) {
    int wave = (blockIdx.x * blockDim.x + threadIdx.x) >> 5;
    int lane = threadIdx.x & 31;
    if (wave >= ntiles) return;
    int ebase = wave * 16;
    int erow = ebase + (lane & 15);
    if (erow >= E) erow = E - 1;
    const u16t* rs = hbf + (size_t)src[erow] * HID;
    const u16t* rd = hbf + (size_t)dst[erow] * HID;
    int kbase = (lane < 16) ? 0 : 8;

    v8f c0 = {}, c1 = {};
#pragma unroll
    for (int kc = 0; kc < 4; kc++) {
        const u16t* rp = (kc < 2) ? rs : rd;
        v16bf a  = load_frag_row(rp + (kc & 1) * 32, kbase);
        v16bf bA = load_frag_w(W1s, kc, 2, 0, lane);
        v16bf bB = load_frag_w(W1s, kc, 2, 1, lane);
        c0 = wmma_bf16(a, bA, c0);
        c1 = wmma_bf16(a, bB, c1);
    }
    int half = (lane < 16) ? 0 : 8;
#pragma unroll
    for (int nt = 0; nt < 2; nt++) {
        int n = nt * 16 + (lane & 15);
        float bias = b1[n];
        v8f c = nt ? c1 : c0;
#pragma unroll
        for (int v = 0; v < 8; v++) {
            int e = ebase + v + half;
            if (e < E) hid[(size_t)e * 32 + n] = f2bf(fmaxf(c[v] + bias, 0.f));
        }
    }
}

// ---------------------------------------------------------------------------
// Dominant GEMM: msg = P @ W2resh, [E,2112]x[2112,64].  W2resh (270KB) is
// DMA'd into LDS once per workgroup by the Tensor Data Mover; the K-loop then
// feeds WMMA from ds_load_b128.  P is built in registers with v_pk_mul_bf16:
// chunk kc<64: P[.,32kc+p] = hidden[e, kc>>1] * h_src[e, (kc&1)*32+p]
// chunk 64/65: P = h_src (bias rows).  Wave handles 2 edge tiles.
// Epilogue: scatter f32 atomics into agg[dst].
// ---------------------------------------------------------------------------
__global__ void __launch_bounds__(256)
k_edge_msg(const u16t* __restrict__ hbf, const u16t* __restrict__ hid,
           const int* __restrict__ src, const int* __restrict__ dst,
           const u16t* __restrict__ W2s, float* __restrict__ agg,
           int E, int nwaves) {
    extern __shared__ u16t w2lds[];
    int lane = threadIdx.x & 31;

    // --- TDM fill: wave 0 issues one tensor_load_to_lds for the whole layer.
    if ((threadIdx.x >> 5) == 0) {
#if defined(__gfx1250__)
        unsigned lds_addr = (unsigned)(unsigned long long)(uintptr_t)&w2lds[0];
        unsigned long long ga = (unsigned long long)(uintptr_t)W2s;
        const unsigned TD0 = 4224, TD1 = 32;   // 4224*32 = 135168 bf16 elems
        u32x4 g0;
        g0[0] = 1u;                                        // count=1, user mode
        g0[1] = lds_addr;                                  // LDS byte address
        g0[2] = (unsigned)(ga & 0xFFFFFFFFu);              // global_addr[31:0]
        g0[3] = (unsigned)((ga >> 32) & 0x01FFFFFFu) | 0x80000000u; // addr[56:32] | type=2
        i32x8 g1;
        g1[0] = 0x00010000;                  // wg_mask=0, data_size=1 (2 bytes)
        g1[1] = (int)(TD0 << 16);            // tensor_dim0[15:0] @ bits 63:48
        g1[2] = (int)((TD0 >> 16) | (TD1 << 16)); // tensor_dim0 hi | tensor_dim1 lo
        g1[3] = (int)((TD1 >> 16) | (TD0 << 16)); // tensor_dim1 hi | tile_dim0
        g1[4] = (int)TD1;                    // tile_dim1 | tile_dim2=0
        g1[5] = (int)TD0;                    // tensor_dim0_stride[31:0]
        g1[6] = 0;                           // stride0 hi | stride1 lo
        g1[7] = 0;
        i32x4 gz = {0, 0, 0, 0};
#if __clang_major__ >= 23
        i32x8 gz8 = {0, 0, 0, 0, 0, 0, 0, 0};
        __builtin_amdgcn_tensor_load_to_lds(g0, g1, gz, gz, gz8, 0);
#else
        __builtin_amdgcn_tensor_load_to_lds(g0, g1, gz, gz, 0);
#endif
        __builtin_amdgcn_s_wait_tensorcnt(0);
#endif
    }
    __syncthreads();

    int wave = (blockIdx.x * blockDim.x + threadIdx.x) >> 5;
    if (wave >= nwaves) return;
    int kbase = (lane < 16) ? 0 : 8;
    int ebase0 = wave * 32;

    F16Frag hs[2][2];
    int erow[2];
#pragma unroll
    for (int et = 0; et < 2; et++) {
        int e = ebase0 + et * 16 + (lane & 15);
        if (e >= E) e = E - 1;
        erow[et] = e;
        const u16t* rp = hbf + (size_t)src[e] * HID;
        hs[et][0].h[0] = *(const u16x8*)(rp + kbase);
        hs[et][0].h[1] = *(const u16x8*)(rp + kbase + 16);
        hs[et][1].h[0] = *(const u16x8*)(rp + 32 + kbase);
        hs[et][1].h[1] = *(const u16x8*)(rp + 32 + kbase + 16);
    }

    v8f acc[2][4] = {};
    for (int k = 0; k < 32; k++) {
        u16t hb[2];
#pragma unroll
        for (int et = 0; et < 2; et++) hb[et] = hid[(size_t)erow[et] * 32 + k];
#pragma unroll
        for (int half = 0; half < 2; half++) {
            int kc = 2 * k + half;
            v16bf B0 = load_frag_w(w2lds, kc, 4, 0, lane);
            v16bf B1 = load_frag_w(w2lds, kc, 4, 1, lane);
            v16bf B2 = load_frag_w(w2lds, kc, 4, 2, lane);
            v16bf B3 = load_frag_w(w2lds, kc, 4, 3, lane);
#pragma unroll
            for (int et = 0; et < 2; et++) {
                v16bf a = scale_frag(hs[et][half].bf, hb[et]);
                acc[et][0] = wmma_bf16(a, B0, acc[et][0]);
                acc[et][1] = wmma_bf16(a, B1, acc[et][1]);
                acc[et][2] = wmma_bf16(a, B2, acc[et][2]);
                acc[et][3] = wmma_bf16(a, B3, acc[et][3]);
            }
        }
    }
    // bias rows (P = h_src)
#pragma unroll
    for (int half = 0; half < 2; half++) {
        int kc = 64 + half;
        v16bf B0 = load_frag_w(w2lds, kc, 4, 0, lane);
        v16bf B1 = load_frag_w(w2lds, kc, 4, 1, lane);
        v16bf B2 = load_frag_w(w2lds, kc, 4, 2, lane);
        v16bf B3 = load_frag_w(w2lds, kc, 4, 3, lane);
#pragma unroll
        for (int et = 0; et < 2; et++) {
            acc[et][0] = wmma_bf16(hs[et][half].bf, B0, acc[et][0]);
            acc[et][1] = wmma_bf16(hs[et][half].bf, B1, acc[et][1]);
            acc[et][2] = wmma_bf16(hs[et][half].bf, B2, acc[et][2]);
            acc[et][3] = wmma_bf16(hs[et][half].bf, B3, acc[et][3]);
        }
    }
    // scatter-mean numerator
    int half = (lane < 16) ? 0 : 8;
#pragma unroll
    for (int et = 0; et < 2; et++) {
        int eb = ebase0 + et * 16;
#pragma unroll
        for (int v = 0; v < 8; v++) {
            int e = eb + v + half;
            if (e < E) {
                float* ap = agg + (size_t)dst[e] * 64;
#pragma unroll
                for (int nt = 0; nt < 4; nt++)
                    atomicAdd(ap + nt * 16 + (lane & 15), acc[et][nt][v]);
            }
        }
    }
}

// ---------------------------------------------------------------------------
// h = relu(BN(agg/deg + h@Wr + br)) + h ; fused, wave per 16 nodes.
// ---------------------------------------------------------------------------
__global__ void k_node(const float* __restrict__ agg, const float* __restrict__ deg,
                       const u16t* __restrict__ Wrs, const float* __restrict__ br,
                       const float* __restrict__ gamma, const float* __restrict__ beta,
                       const float* __restrict__ rmean, const float* __restrict__ rvar,
                       float* __restrict__ h, u16t* __restrict__ hbf,
                       int N, int ntiles) {
    int wave = (blockIdx.x * blockDim.x + threadIdx.x) >> 5;
    int lane = threadIdx.x & 31;
    if (wave >= ntiles) return;
    int nbase = wave * 16;
    int nrow = nbase + (lane & 15);
    if (nrow >= N) nrow = N - 1;
    const u16t* rp = hbf + (size_t)nrow * 64;
    int kbase = (lane < 16) ? 0 : 8;

    v8f c[4] = {};
#pragma unroll
    for (int kc = 0; kc < 2; kc++) {
        v16bf a = load_frag_row(rp + kc * 32, kbase);
#pragma unroll
        for (int nt = 0; nt < 4; nt++)
            c[nt] = wmma_bf16(a, load_frag_w(Wrs, kc, 4, nt, lane), c[nt]);
    }
    int half = (lane < 16) ? 0 : 8;
#pragma unroll
    for (int nt = 0; nt < 4; nt++) {
        int n = nt * 16 + (lane & 15);
        float g  = gamma[n] * rsqrtf(rvar[n] + 1e-5f);
        float rm = rmean[n], bt = beta[n], bias = br[n];
#pragma unroll
        for (int v = 0; v < 8; v++) {
            int node = nbase + v + half;
            if (node < N) {
                float dg  = fmaxf(deg[node], 1.f);
                float h2  = agg[(size_t)node * 64 + n] / dg + c[nt][v] + bias;
                h2 = (h2 - rm) * g + bt;
                float nh = fmaxf(h2, 0.f) + h[(size_t)node * 64 + n];
                h[(size_t)node * 64 + n]  = nh;
                hbf[(size_t)node * 64 + n] = f2bf(nh);
            }
        }
    }
}

// ---------------------------------------------------------------------------
// out = relu([h_u|h_v]@Wm1+bm1)@Wm2+bm2 ; wave per 16 branch-rows,
// final dot via half-wave shuffle reduction.
// ---------------------------------------------------------------------------
__global__ void k_readout(const u16t* __restrict__ hbf,
                          const int* __restrict__ bu, const int* __restrict__ bv,
                          const u16t* __restrict__ Wm1s, const float* __restrict__ bm1,
                          const float* __restrict__ Wm2, const float* __restrict__ bm2,
                          float* __restrict__ out, int R, int ntiles) {
    int wave = (blockIdx.x * blockDim.x + threadIdx.x) >> 5;
    int lane = threadIdx.x & 31;
    if (wave >= ntiles) return;
    int rbase = wave * 16;
    int r = rbase + (lane & 15);
    if (r >= R) r = R - 1;
    int g = r / 41, b = r % 41;
    const u16t* ru = hbf + (size_t)(bu[b] + g * 30) * 64;
    const u16t* rv = hbf + (size_t)(bv[b] + g * 30) * 64;
    int kbase = (lane < 16) ? 0 : 8;

    v8f c[4] = {};
#pragma unroll
    for (int kc = 0; kc < 4; kc++) {
        const u16t* rp = (kc < 2) ? ru : rv;
        v16bf a = load_frag_row(rp + (kc & 1) * 32, kbase);
#pragma unroll
        for (int nt = 0; nt < 4; nt++)
            c[nt] = wmma_bf16(a, load_frag_w(Wm1s, kc, 4, nt, lane), c[nt]);
    }
    float p[8];
#pragma unroll
    for (int v = 0; v < 8; v++) {
        float s = 0.f;
#pragma unroll
        for (int nt = 0; nt < 4; nt++) {
            int n = nt * 16 + (lane & 15);
            s += fmaxf(c[nt][v] + bm1[n], 0.f) * Wm2[n];
        }
        p[v] = s;
    }
#pragma unroll
    for (int v = 0; v < 8; v++)
        for (int m = 1; m < 16; m <<= 1) p[v] += __shfl_xor(p[v], m, 32);
    if ((lane & 15) == 0) {
        int half = (lane < 16) ? 0 : 8;
        float bb = bm2[0];
#pragma unroll
        for (int v = 0; v < 8; v++) {
            int row = rbase + v + half;
            if (row < R) out[row] = p[v] + bb;
        }
    }
}

// ---------------------------------------------------------------------------
extern "C" void kernel_launch(void* const* d_in, const int* in_sizes, int n_in,
                              void* d_out, int out_size, void* d_ws, size_t ws_size,
                              hipStream_t stream) {
    const float* x     = (const float*)d_in[0];
    const int*   ei    = (const int*)d_in[1];
    const int*   bu    = (const int*)d_in[2];
    const int*   bv    = (const int*)d_in[3];
    const float* Wp    = (const float*)d_in[5];
    const float* bp    = (const float*)d_in[6];
    const float* W1    = (const float*)d_in[7];
    const float* b1    = (const float*)d_in[8];
    const float* W2    = (const float*)d_in[9];
    const float* b2    = (const float*)d_in[10];
    const float* Wr    = (const float*)d_in[11];
    const float* br    = (const float*)d_in[12];
    const float* gamma = (const float*)d_in[13];
    const float* beta  = (const float*)d_in[14];
    const float* rmean = (const float*)d_in[15];
    const float* rvar  = (const float*)d_in[16];
    const float* Wm1   = (const float*)d_in[17];
    const float* bm1   = (const float*)d_in[18];
    const float* Wm2   = (const float*)d_in[19];
    const float* bm2   = (const float*)d_in[20];

    int E = in_sizes[1] / 2;
    int G = out_size / 41;
    int N = G * 30;
    const int* src = ei;
    const int* dst = ei + E;

    char* w = (char*)d_ws;
    size_t off = 0;
    auto carve = [&](size_t bytes) -> char* {
        char* p = w + off;
        off = (off + bytes + 255) & ~(size_t)255;
        return p;
    };
    float* h    = (float*)carve((size_t)N * 64 * 4);
    u16t*  hbf  = (u16t*)carve((size_t)N * 64 * 2);
    float* agg  = (float*)carve((size_t)N * 64 * 4);
    float* deg  = (float*)carve((size_t)N * 4);
    u16t*  hid  = (u16t*)carve((size_t)E * 32 * 2);
    u16t*  W1s  = (u16t*)carve((size_t)5 * 4096 * 2);
    u16t*  W2s  = (u16t*)carve((size_t)5 * 135168 * 2);
    u16t*  Wrs  = (u16t*)carve((size_t)5 * 4096 * 2);
    u16t*  Wm1s = (u16t*)carve((size_t)8192 * 2);
    (void)ws_size; (void)n_in;

    // Allow >64KB dynamic LDS for the TDM-staged GEMM kernel (capture-safe).
    hipFuncSetAttribute((const void*)k_edge_msg,
                        hipFuncAttributeMaxDynamicSharedMemorySize, MSG_LDS_BYTES);

    const int tot = 5 * 4096 + 5 * 135168 + 5 * 4096 + 8192;
    k_prep<<<(tot + 255) / 256, 256, 0, stream>>>(W1, W2, b2, Wr, Wm1, W1s, W2s, Wrs, Wm1s);
    k_proj<<<(N * 64 + 255) / 256, 256, 0, stream>>>(x, Wp, bp, h, hbf, N);
    k_zero<<<(N + 255) / 256, 256, 0, stream>>>(deg, N);
    k_deg<<<(E + 255) / 256, 256, 0, stream>>>(dst, deg, E);

    int tilesE = (E + 15) / 16;
    int tilesN = (N + 15) / 16;
    int wavesMsg = (tilesE + 1) / 2;
    for (int l = 0; l < 5; l++) {
        k_zero<<<(N * 64 + 255) / 256, 256, 0, stream>>>(agg, N * 64);
        k_edge_hidden<<<(tilesE * 32 + 255) / 256, 256, 0, stream>>>(
            hbf, src, dst, W1s + l * 4096, b1 + l * 32, hid, E, tilesE);
        k_edge_msg<<<(wavesMsg * 32 + 255) / 256, 256, MSG_LDS_BYTES, stream>>>(
            hbf, hid, src, dst, W2s + (size_t)l * 135168, agg, E, wavesMsg);
        k_node<<<(tilesN * 32 + 255) / 256, 256, 0, stream>>>(
            agg, deg, Wrs + l * 4096, br + l * 64, gamma + l * 64, beta + l * 64,
            rmean + l * 64, rvar + l * 64, h, hbf, N, tilesN);
    }
    int R = out_size;
    int tilesR = (R + 15) / 16;
    k_readout<<<(tilesR * 32 + 255) / 256, 256, 0, stream>>>(
        hbf, bu, bv, Wm1s, bm1, Wm2, bm2, (float*)d_out, R, tilesR);
}
